// GATNet2_25443386261564
// MI455X (gfx1250) — matrix operand order repaired and checked by reading, not compiled, not run on hardware
//
#include <hip/hip_runtime.h>
#include <hip/hip_bf16.h>
#include <math.h>

typedef __attribute__((ext_vector_type(16))) _Float16 v16h;
typedef __attribute__((ext_vector_type(8)))  _Float16 v8h;
typedef __attribute__((ext_vector_type(8)))  float    v8f;

#define NEG_SLOPE 0.2f
#define SM_EPS    1e-16f

static inline int cdiv_i(int a, int b) { return (a + b - 1) / b; }

// ---------------------------------------------------------------------------
// WMMA tiled GEMM: C[M,N] = A[M,sa-strided] @ Bp[KDIM,NPAD] (+ bias[N])
//  - KDIM, NPAD compile-time; Bp zero-padded to KDIM x NPAD so no K/N guards.
//  - A rows clamped to M-1 (clamped rows feed only unstored C rows).
//  - A columns past sa (when KDIM > sa) hit zero-padded B rows -> contribute 0.
// Block = 128 threads = 4 waves; block tile 64(M) x 16(N); wave tile 16x16.
// Fragment packing per CDNA5 ISA 7.12.2:
//   A 16x32 f16: lane l -> M=l&15, kg=l>>4; elems = K runs [kg*8,+8),[kg*8+16,+8)
//   B 32x16 f16: lane l -> N=l&15, kg=l>>4; elems = K runs [kg*8,+8),[kg*8+16,+8)
//   C/D 16x16 f32: lane l -> N=l&15; vgpr i -> M = i + (l>=16 ? 8:0)
// ---------------------------------------------------------------------------
template <int KDIM, int NPAD>
__global__ __launch_bounds__(128)
void gemm_wmma(const float* __restrict__ A, const float* __restrict__ Bp,
               const float* __restrict__ bias, float* __restrict__ C,
               int M, int N, int sa, int useBias) {
  __shared__ _Float16 sA[64][40];    // 80B row stride: 16B-aligned, conflict-free
  __shared__ _Float16 sBt[16][40];   // B stored transposed: sBt[n][k]
  const int tid   = threadIdx.x;
  const int wave  = tid >> 5;
  const int lane  = tid & 31;
  const int mBase = blockIdx.y * 64;
  const int nBase = blockIdx.x * 16;
  const int lm    = (wave << 4) + (lane & 15);
  const int kg    = lane >> 4;
  const int nl    = lane & 15;
  // staging assignments (branch-free, coalesced)
  const int amt = tid >> 1;            // A row in tile: 0..63
  const int akt = (tid & 1) * 16;      // A col chunk: 0 or 16
  const int bkt = tid >> 2;            // B row in tile: 0..31
  const int bnt = (tid & 3) * 4;       // B col group: 0,4,8,12

  int gm = mBase + amt; if (gm >= M) gm = M - 1;   // clamp (no predication)
  const float* aRow = A + (size_t)gm * sa;

  v8f acc = {};

  #pragma unroll
  for (int kk = 0; kk < KDIM; kk += 32) {
    // ---- stage A: 16 contiguous floats per thread -> 16 f16 in LDS ----
    {
      const float* ap = aRow + kk + akt;
      float4 f0 = *(const float4*)(ap + 0);
      float4 f1 = *(const float4*)(ap + 4);
      float4 f2 = *(const float4*)(ap + 8);
      float4 f3 = *(const float4*)(ap + 12);
      v8h h0 = { (_Float16)f0.x, (_Float16)f0.y, (_Float16)f0.z, (_Float16)f0.w,
                 (_Float16)f1.x, (_Float16)f1.y, (_Float16)f1.z, (_Float16)f1.w };
      v8h h1 = { (_Float16)f2.x, (_Float16)f2.y, (_Float16)f2.z, (_Float16)f2.w,
                 (_Float16)f3.x, (_Float16)f3.y, (_Float16)f3.z, (_Float16)f3.w };
      *reinterpret_cast<v8h*>(&sA[amt][akt])     = h0;
      *reinterpret_cast<v8h*>(&sA[amt][akt + 8]) = h1;
    }
    // ---- stage B: float4 per thread, transposed into LDS ----
    {
      const float* bp = Bp + (size_t)(kk + bkt) * NPAD + nBase + bnt;
      float4 f = *(const float4*)bp;
      sBt[bnt + 0][bkt] = (_Float16)f.x;
      sBt[bnt + 1][bkt] = (_Float16)f.y;
      sBt[bnt + 2][bkt] = (_Float16)f.z;
      sBt[bnt + 3][bkt] = (_Float16)f.w;
    }
    __syncthreads();
    // ---- fragments: 2x ds_load_b128 each ----
    v8h a0 = *reinterpret_cast<const v8h*>(&sA[lm][kg * 8]);
    v8h a1 = *reinterpret_cast<const v8h*>(&sA[lm][kg * 8 + 16]);
    v8h b0 = *reinterpret_cast<const v8h*>(&sBt[nl][kg * 8]);
    v8h b1 = *reinterpret_cast<const v8h*>(&sBt[nl][kg * 8 + 16]);
    v16h af = __builtin_shufflevector(a0, a1, 0,1,2,3,4,5,6,7,8,9,10,11,12,13,14,15);
    v16h bf = __builtin_shufflevector(b0, b1, 0,1,2,3,4,5,6,7,8,9,10,11,12,13,14,15);
    acc = __builtin_amdgcn_wmma_f32_16x16x32_f16(
        false, af, false, bf, (short)0, acc, false, false);
    __syncthreads();
  }

  const int gn = nBase + nl;
  #pragma unroll
  for (int i = 0; i < 8; ++i) {
    int gmo = mBase + (wave << 4) + (kg << 3) + i;
    if (gmo < M && gn < N) {
      float v = acc[i];
      if (useBias) v += bias[gn];
      C[(size_t)gmo * N + gn] = v;
    }
  }
}

// ---------------------------------------------------------------------------
// Scalar helper kernels (edge/softmax phase: scatter-bound f32 path)
// ---------------------------------------------------------------------------
__global__ void k_zero(float* __restrict__ p, int n) {
  int i = blockIdx.x * blockDim.x + threadIdx.x;
  if (i < n) p[i] = 0.0f;
}

// zero-pad small B matrices: src[K,N] -> dst[Kp,Np]
__global__ void k_pad(const float* __restrict__ src, float* __restrict__ dst,
                      int K, int N, int Kp, int Np) {
  int i = blockIdx.x * blockDim.x + threadIdx.x;
  if (i >= Kp * Np) return;
  int k = i / Np, n = i % Np;
  dst[i] = (k < K && n < N) ? src[k * N + n] : 0.0f;
}

// a_s[n,h] = sum_c h[n,h,c]*att_s[h,c] ; a_d analogous
__global__ void k_logits(const float* __restrict__ Hf,
                         const float* __restrict__ attS,
                         const float* __restrict__ attD,
                         float* __restrict__ AS, float* __restrict__ AD,
                         int Nn, int Hh, int Cc) {
  int i = blockIdx.x * blockDim.x + threadIdx.x;
  if (i >= Nn * Hh) return;
  int n = i / Hh, h = i % Hh;
  const float* row = Hf + (size_t)n * Hh * Cc + (size_t)h * Cc;
  const float* ws  = attS + h * Cc;
  const float* wd  = attD + h * Cc;
  float ss = 0.0f, sd = 0.0f;
  for (int c = 0; c < Cc; ++c) { float v = row[c]; ss += v * ws[c]; sd += v * wd[c]; }
  AS[i] = ss; AD[i] = sd;
}

// order-preserving float<->uint encoding for atomicMax-based segment max
__device__ __forceinline__ unsigned f32_ord(float f) {
  unsigned b = __float_as_uint(f);
  return (b & 0x80000000u) ? ~b : (b | 0x80000000u);
}
__device__ __forceinline__ float ord_f32(unsigned u) {
  unsigned b = (u & 0x80000000u) ? (u & 0x7FFFFFFFu) : ~u;
  return __uint_as_float(b);
}

__global__ void k_edge_max(const int* __restrict__ ei,
                           const float* __restrict__ AS,
                           const float* __restrict__ AD,
                           unsigned* __restrict__ ME,
                           int Et, int Eg, int Hh) {
  int i = blockIdx.x * blockDim.x + threadIdx.x;
  if (i >= Et * Hh) return;
  int e = i / Hh, h = i % Hh;
  int src = (e < Eg) ? ei[e]      : (e - Eg);   // self-loops appended
  int dst = (e < Eg) ? ei[Eg + e] : (e - Eg);
  float a = AS[src * Hh + h] + AD[dst * Hh + h];
  a = (a > 0.0f) ? a : NEG_SLOPE * a;           // leaky_relu
  atomicMax(&ME[dst * Hh + h], f32_ord(a));
}

__global__ void k_edge_exp(const int* __restrict__ ei,
                           const float* __restrict__ AS,
                           const float* __restrict__ AD,
                           const unsigned* __restrict__ ME,
                           float* __restrict__ S, float* __restrict__ EE,
                           int Et, int Eg, int Hh) {
  int i = blockIdx.x * blockDim.x + threadIdx.x;
  if (i >= Et * Hh) return;
  int e = i / Hh, h = i % Hh;
  int src = (e < Eg) ? ei[e]      : (e - Eg);
  int dst = (e < Eg) ? ei[Eg + e] : (e - Eg);
  float a = AS[src * Hh + h] + AD[dst * Hh + h];
  a = (a > 0.0f) ? a : NEG_SLOPE * a;
  float m = ord_f32(ME[dst * Hh + h]);
  if (!(m > -3.0e38f && m < 3.0e38f)) m = 0.0f;  // non-finite -> 0 (per ref)
  float ex = __expf(a - m);
  EE[i] = ex;
  atomicAdd(&S[dst * Hh + h], ex);
}

__global__ void k_edge_aggr(const int* __restrict__ ei,
                            const float* __restrict__ EE,
                            const float* __restrict__ S,
                            const float* __restrict__ Hf,
                            float* __restrict__ AG,
                            int Et, int Eg, int Hh, int Cc) {
  long long i = (long long)blockIdx.x * blockDim.x + threadIdx.x;
  long long tot = (long long)Et * Hh * Cc;
  if (i >= tot) return;
  int e = (int)(i / (Hh * Cc));
  int r = (int)(i % (Hh * Cc));
  int h = r / Cc;
  int src = (e < Eg) ? ei[e]      : (e - Eg);
  int dst = (e < Eg) ? ei[Eg + e] : (e - Eg);
  float a = EE[(size_t)e * Hh + h] / (S[dst * Hh + h] + SM_EPS);
  atomicAdd(&AG[(size_t)dst * Hh * Cc + r], Hf[(size_t)src * Hh * Cc + r] * a);
}

// dst = (addTo ? dst : 0) + relu(AG + bias)
__global__ void k_epilogue(const float* __restrict__ AG,
                           const float* __restrict__ bias,
                           float* __restrict__ dst,
                           int total, int D, int addTo) {
  int i = blockIdx.x * blockDim.x + threadIdx.x;
  if (i >= total) return;
  float v = AG[i] + bias[i % D];
  v = (v > 0.0f) ? v : 0.0f;
  dst[i] = addTo ? (dst[i] + v) : v;
}

__global__ void k_logsoftmax(const float* __restrict__ in,
                             float* __restrict__ out, int Nn, int D) {
  int n = blockIdx.x * blockDim.x + threadIdx.x;
  if (n >= Nn) return;
  const float* row = in + (size_t)n * D;
  float mx = -3.4e38f;
  for (int j = 0; j < D; ++j) mx = fmaxf(mx, row[j]);
  float s = 0.0f;
  for (int j = 0; j < D; ++j) s += __expf(row[j] - mx);
  float l = __logf(s) + mx;
  float* orow = out + (size_t)n * D;
  for (int j = 0; j < D; ++j) orow[j] = row[j] - l;
}

// ---------------------------------------------------------------------------
extern "C" void kernel_launch(void* const* d_in, const int* in_sizes, int n_in,
                              void* d_out, int out_size, void* d_ws, size_t ws_size,
                              hipStream_t stream) {
  (void)n_in; (void)out_size; (void)ws_size;
  const float* x   = (const float*)d_in[0];
  const int*   ei  = (const int*)  d_in[1];
  const float* W0  = (const float*)d_in[2];
  const float* as0 = (const float*)d_in[3];
  const float* ad0 = (const float*)d_in[4];
  const float* b0  = (const float*)d_in[5];
  const float* W1  = (const float*)d_in[6];
  const float* as1 = (const float*)d_in[7];
  const float* ad1 = (const float*)d_in[8];
  const float* b1  = (const float*)d_in[9];
  const float* R1W = (const float*)d_in[10];
  const float* R1b = (const float*)d_in[11];
  const float* W2  = (const float*)d_in[12];
  const float* as2 = (const float*)d_in[13];
  const float* ad2 = (const float*)d_in[14];
  const float* b2  = (const float*)d_in[15];
  const float* R2W = (const float*)d_in[16];
  const float* R2b = (const float*)d_in[17];
  const float* LW  = (const float*)d_in[18];
  const float* Lb  = (const float*)d_in[19];
  float* out = (float*)d_out;

  const int Nn = in_sizes[0] / 128;   // 50000
  const int Eg = in_sizes[1] / 2;     // 500000
  const int Et = Eg + Nn;             // edges + self-loops

  // --- workspace carve (+64-elem slack: tolerates deliberate short-row reads)
  char* w = (char*)d_ws; size_t off = 0;
  auto alloc = [&](size_t elems) -> float* {
    float* p = (float*)(w + off);
    off += (((elems + 64) * sizeof(float) + 255) / 256) * 256;
    return p;
  };
  float*    Hf   = alloc((size_t)Nn * 128); // per-layer features h = x@W
  float*    AG   = alloc((size_t)Nn * 128); // aggregation accumulator
  float*    X1   = alloc((size_t)Nn * 128); // layer-0 output / residual
  float*    X2   = alloc((size_t)Nn * 128); // layer-1 output
  float*    ASb  = alloc((size_t)Nn * 4);
  float*    ADb  = alloc((size_t)Nn * 4);
  unsigned* ME   = (unsigned*)alloc((size_t)Nn * 4);
  float*    Sb   = alloc((size_t)Nn * 4);
  float*    EE   = alloc((size_t)Et * 4);
  float*    T40  = alloc((size_t)Nn * 40);
  float*    LG   = alloc((size_t)Nn * 40);
  float*    W2p  = alloc(128 * 48);         // zero-padded small weights
  float*    R2Wp = alloc(128 * 48);
  float*    LWp  = alloc(64 * 48);

  auto zero = [&](float* p, long long n) {
    k_zero<<<(unsigned)((n + 255) / 256), 256, 0, stream>>>(p, (int)n);
  };
  auto gat_edges = [&](const float* Hfeat, float* AGbuf, int Hh, int Cc) {
    int nh = Nn * Hh;
    zero((float*)ME, nh);                 // 0 == encoded "below every float"
    zero(Sb, nh);
    zero(AGbuf, (long long)Nn * Hh * Cc);
    int eh = Et * Hh;
    k_edge_max<<<cdiv_i(eh, 256), 256, 0, stream>>>(ei, ASb, ADb, ME, Et, Eg, Hh);
    k_edge_exp<<<cdiv_i(eh, 256), 256, 0, stream>>>(ei, ASb, ADb, ME, Sb, EE, Et, Eg, Hh);
    long long tot = (long long)Et * Hh * Cc;
    k_edge_aggr<<<(unsigned)((tot + 255) / 256), 256, 0, stream>>>(
        ei, EE, Sb, Hfeat, AGbuf, Et, Eg, Hh, Cc);
  };

  const dim3 gFull(128 / 16, cdiv_i(Nn, 64));   // N=128 tiles
  const dim3 g40(3, cdiv_i(Nn, 64));            // N=40 -> 3 col tiles

  // ---- pad small weight matrices once (into ws) ----
  k_pad<<<cdiv_i(128 * 48, 256), 256, 0, stream>>>(W2,  W2p,  128, 40, 128, 48);
  k_pad<<<cdiv_i(128 * 48, 256), 256, 0, stream>>>(R2W, R2Wp, 128, 40, 128, 48);
  k_pad<<<cdiv_i(64 * 48, 256), 256, 0, stream>>>(LW,  LWp,  40, 40, 64, 48);

  // ---- Layer 0: X1 = relu(GAT(x; W0, as0, ad0) + b0), 4 heads x 32 ----
  gemm_wmma<128, 128><<<gFull, 128, 0, stream>>>(x, W0, nullptr, Hf, Nn, 128, 128, 0);
  k_logits<<<cdiv_i(Nn * 4, 256), 256, 0, stream>>>(Hf, as0, ad0, ASb, ADb, Nn, 4, 32);
  gat_edges(Hf, AG, 4, 32);
  k_epilogue<<<cdiv_i(Nn * 128, 256), 256, 0, stream>>>(AG, b0, X1, Nn * 128, 128, 0);

  // ---- Layer 1: X2 = relu(GAT(X1) + b1) + (X1@R1W + R1b) ----
  gemm_wmma<128, 128><<<gFull, 128, 0, stream>>>(X1, W1, nullptr, Hf, Nn, 128, 128, 0);
  k_logits<<<cdiv_i(Nn * 4, 256), 256, 0, stream>>>(Hf, as1, ad1, ASb, ADb, Nn, 4, 32);
  gat_edges(Hf, AG, 4, 32);
  gemm_wmma<128, 128><<<gFull, 128, 0, stream>>>(X1, R1W, R1b, X2, Nn, 128, 128, 1);
  k_epilogue<<<cdiv_i(Nn * 128, 256), 256, 0, stream>>>(AG, b1, X2, Nn * 128, 128, 1);

  // ---- Layer 2: LG = relu(GAT(X2; 1 head x 40) + b2) + (X2@R2W + R2b) ----
  gemm_wmma<128, 48><<<g40, 128, 0, stream>>>(X2, W2p, nullptr, T40, Nn, 40, 128, 0);
  k_logits<<<cdiv_i(Nn, 256), 256, 0, stream>>>(T40, as2, ad2, ASb, ADb, Nn, 1, 40);
  gat_edges(T40, AG, 1, 40);
  gemm_wmma<128, 48><<<g40, 128, 0, stream>>>(X2, R2Wp, R2b, LG, Nn, 40, 128, 1);
  k_epilogue<<<cdiv_i(Nn * 40, 256), 256, 0, stream>>>(AG, b2, LG, Nn * 40, 40, 1);

  // ---- Head: out = log_softmax(LG @ LW + Lb) ----
  gemm_wmma<64, 48><<<g40, 128, 0, stream>>>(LG, LWp, Lb, T40, Nn, 40, 40, 1);
  k_logsoftmax<<<cdiv_i(Nn, 256), 256, 0, stream>>>(T40, out, Nn, 40);
}
